// GNN_55679956025643
// MI455X (gfx1250) — matrix operand order repaired and checked by reading, not compiled, not run on hardware
//
#include <hip/hip_runtime.h>
#include <stdint.h>

// ---------------- CDNA5 WMMA types ----------------
typedef __attribute__((ext_vector_type(16))) __bf16 v16bf;
typedef __attribute__((ext_vector_type(8)))  float  v8f;

#define U_N     100000
#define I_NODES 100000
#define D_DIM   128
#define E_N     200000
#define EA_DIM  768

__device__ __forceinline__ unsigned short f2bf(float f) {
  unsigned int u = __float_as_uint(f);
  unsigned int r = u + 0x7FFFu + ((u >> 16) & 1u);   // round-to-nearest-even
  return (unsigned short)(r >> 16);
}
__device__ __forceinline__ float bf2f(unsigned short h) {
  return __uint_as_float(((unsigned int)h) << 16);
}
__device__ __forceinline__ unsigned int pack2(float lo, float hi) {
  return (unsigned int)f2bf(lo) | ((unsigned int)f2bf(hi) << 16);
}

// CDNA5 async global->LDS copy (ASYNCcnt-tracked, bypasses VGPRs).
__device__ __forceinline__ void async_copy_b128(unsigned lds_addr, const void* gaddr) {
  asm volatile("global_load_async_to_lds_b128 %0, %1, off"
               :: "v"(lds_addr), "v"(gaddr) : "memory");
}
__device__ __forceinline__ void async_wait0() {
  asm volatile("s_wait_asynccnt 0x0" ::: "memory");
}

// ---------------- generic WMMA GEMM ----------------
// C[M,N] = epilogue(A[M,K] @ B[K,N] + bias)
// AMODE 0: A bf16 (async-staged)   AMODE 1: A fp32, converted on the fly
// AMODE 2: per-edge concat gather of two bf16 node tables (K==256)
// BT: B transposed, bf16 [N][K] (async-staged).
// CMODE 0: store f32   CMODE 1: store bf16
// CMODE 2: msg = relu(acc+bias) + gather(xj) ; atomicAdd into agg[dst]  (N==128)
// CMODE 3: out = relu(acc+bias + agg[m]/max(cnt,1)) -> bf16              (N==128)
#define BM 128
#define BN 128
#define KC 128
#define NTHREADS 256
#define LSTRIDE 68        // dword row stride: 16B-aligned, shifts banks by 4/row

template<int AMODE, int CMODE, bool RELU>
__global__ __launch_bounds__(NTHREADS)
void wmma_gemm(const void* __restrict__ Aptr,
               const unsigned short* __restrict__ BT,
               const float* __restrict__ bias,
               void* __restrict__ Cptr,
               int M, int N, int K,
               const int* __restrict__ g_ei0, const int* __restrict__ g_ei1,
               const unsigned short* __restrict__ g_xu, const unsigned short* __restrict__ g_xi,
               const int* __restrict__ s_src, const int* __restrict__ s_dst,
               const int* __restrict__ s_et,  const unsigned short* __restrict__ s_xj,
               float* __restrict__ s_agg,
               const float* __restrict__ f_agg, const int* __restrict__ f_cnt)
{
  // double-buffered tiles: packed bf16 pairs, row-major [rows][K]
  __shared__ unsigned int As[2][BM * LSTRIDE];
  __shared__ unsigned int Bs[2][BN * LSTRIDE];

  const int tid  = threadIdx.x;
  const int m0   = blockIdx.x * BM;
  const int n0   = blockIdx.y * BN;
  const int wave = tid >> 5;
  const int lane = tid & 31;
  const int grp  = lane >> 4;     // lane group 0/1
  const int lm   = lane & 15;
  const int wm   = wave & 3;      // 4 M-groups of 32 rows
  const int wn   = wave >> 2;     // 2 N-groups of 64 cols

  v8f acc[2][4];
  #pragma unroll
  for (int s = 0; s < 2; ++s)
    #pragma unroll
    for (int t = 0; t < 4; ++t)
      acc[s][t] = v8f{};

  // ---- staging helpers ----
  auto stageA = [&](int buf, int kc) {
    if (AMODE == 0) {
      unsigned base = (unsigned)(unsigned long long)(uintptr_t)&As[buf][0];
      #pragma unroll
      for (int it = 0; it < (BM * KC / 8) / NTHREADS; ++it) {   // 2048/256 = 8
        int q = tid + it * NTHREADS;
        int r = q >> 4;               // row
        int j = q & 15;               // 16B chunk within row
        const char* g = (const char*)Aptr +
            (((size_t)(m0 + r) * (size_t)K + (size_t)kc) << 1) + 16 * j;
        async_copy_b128(base + (unsigned)(r * LSTRIDE + 4 * j) * 4u, g);
      }
    } else {
      for (int idx = tid; idx < BM * (KC / 2); idx += NTHREADS) {
        int r = idx >> 6;             // KC/2 == 64
        int c = idx & 63;
        int m = m0 + r;
        unsigned int val = 0u;
        if (m < M) {
          if (AMODE == 1) {
            const float* Af = (const float*)Aptr;
            size_t b = (size_t)m * (size_t)K + (size_t)kc + 2 * c;
            val = pack2(Af[b], Af[b + 1]);
          } else {                    // AMODE 2: concat gather, K==256
            int k2 = (kc >> 1) + c;
            if (k2 < 64) {
              int nu = g_ei0[m];
              val = ((const unsigned int*)g_xu)[(size_t)nu * 64 + k2];
            } else {
              int ni = g_ei1[m];
              val = ((const unsigned int*)g_xi)[(size_t)ni * 64 + (k2 - 64)];
            }
          }
        }
        As[buf][r * LSTRIDE + c] = val;
      }
    }
  };
  auto stageB = [&](int buf, int kc) {
    unsigned base = (unsigned)(unsigned long long)(uintptr_t)&Bs[buf][0];
    #pragma unroll
    for (int it = 0; it < (BN * KC / 8) / NTHREADS; ++it) {     // 8
      int q = tid + it * NTHREADS;
      int r = q >> 4;
      int j = q & 15;
      const char* g = (const char*)BT +
          (((size_t)(n0 + r) * (size_t)K + (size_t)kc) << 1) + 16 * j;
      async_copy_b128(base + (unsigned)(r * LSTRIDE + 4 * j) * 4u, g);
    }
  };
  auto compute = [&](int buf) {
    #pragma unroll
    for (int kk = 0; kk < KC / 32; ++kk) {
      union F { v16bf v; unsigned int u[8]; };
      F a[2], b[4];
      #pragma unroll
      for (int s = 0; s < 2; ++s) {
        #pragma unroll
        for (int v = 0; v < 8; ++v) {
          // 16-bit A 16x32 layout: lanes0-15 K{0..7,16..23}, lanes16-31 K{8..15,24..31}
          int kOff = kk * 32 + ((v < 4) ? (grp * 8 + 2 * v) : (16 + grp * 8 + 2 * (v - 4)));
          a[s].u[v] = As[buf][(wm * 32 + s * 16 + lm) * LSTRIDE + (kOff >> 1)];
        }
      }
      #pragma unroll
      for (int t = 0; t < 4; ++t) {
        #pragma unroll
        for (int v = 0; v < 8; ++v) {
          // B 32x16 layout: lanes0-15 K{0..7,16..23}, lanes16-31 K{8..15,24..31}, 2 K per VGPR
          int kOff = kk * 32 + ((v < 4) ? (grp * 8 + 2 * v) : (16 + grp * 8 + 2 * (v - 4)));
          b[t].u[v] = Bs[buf][(wn * 64 + t * 16 + lm) * LSTRIDE + (kOff >> 1)];
        }
      }
      #pragma unroll
      for (int t = 0; t < 4; ++t) {
        acc[0][t] = __builtin_amdgcn_wmma_f32_16x16x32_bf16(
            false, a[0].v, false, b[t].v, (short)0, acc[0][t], false, false);
        acc[1][t] = __builtin_amdgcn_wmma_f32_16x16x32_bf16(
            false, a[1].v, false, b[t].v, (short)0, acc[1][t], false, false);
      }
    }
  };

  // ---- double-buffered main loop: prefetch chunk k+1 while computing chunk k ----
  stageA(0, 0);
  stageB(0, 0);
  async_wait0();
  __syncthreads();
  int buf = 0;
  for (int kc = 0; kc < K; kc += KC) {
    const int nkc = kc + KC;
    if (nkc < K) { stageA(buf ^ 1, nkc); stageB(buf ^ 1, nkc); }
    compute(buf);
    if (nkc < K) async_wait0();
    __syncthreads();
    buf ^= 1;
  }

  // ---- epilogue ----
  #pragma unroll
  for (int s = 0; s < 2; ++s) {
    #pragma unroll
    for (int t = 0; t < 4; ++t) {
      int n = n0 + wn * 64 + t * 16 + lm;
      float bv = bias ? bias[n] : 0.f;
      #pragma unroll
      for (int v = 0; v < 8; ++v) {
        int m = m0 + wm * 32 + s * 16 + v + 8 * grp;  // C/D layout: M = vgpr + 8*group
        if (m < M) {
          float x = acc[s][t][v] + bv;
          if (RELU) x = fmaxf(x, 0.f);
          if (CMODE == 0) {
            ((float*)Cptr)[(size_t)m * N + n] = x;
          } else if (CMODE == 1) {
            ((unsigned short*)Cptr)[(size_t)m * N + n] = f2bf(x);
          } else if (CMODE == 2) {
            int src = s_src[m], dst = s_dst[m], et = s_et[m];
            float mn = bf2f(s_xj[(size_t)src * (5 * D_DIM) + (size_t)(et - 1) * D_DIM + n]);
            atomicAdd(&s_agg[(size_t)dst * D_DIM + n], x + mn);
          } else {   // CMODE 3: finalize conv
            int c = f_cnt[m];
            float inv = 1.f / (float)(c > 0 ? c : 1);
            float val = acc[s][t][v] + bv + f_agg[(size_t)m * D_DIM + n] * inv;
            val = fmaxf(val, 0.f);
            ((unsigned short*)Cptr)[(size_t)m * D_DIM + n] = f2bf(val);
          }
        }
      }
    }
  }
}

// ---------------- small helper kernels ----------------
__global__ void transpose_bf16(const float* __restrict__ W, unsigned short* __restrict__ WT,
                               int K, int N) {
  int i = blockIdx.x * blockDim.x + threadIdx.x;
  if (i < K * N) {
    int k = i / N, n = i - k * N;
    WT[(size_t)n * K + k] = f2bf(W[i]);
  }
}

__global__ void count_edges(const int* __restrict__ idx, int* __restrict__ cnt, int E) {
  int i = blockIdx.x * blockDim.x + threadIdx.x;
  if (i < E) atomicAdd(&cnt[idx[i]], 1);
}

__global__ void head_final(const unsigned short* __restrict__ h2,
                           const float* __restrict__ Wr3, const float* __restrict__ br3,
                           float* __restrict__ out, int E) {
  int e = blockIdx.x * blockDim.x + threadIdx.x;
  if (e >= E) return;
  float s = br3[0];
  const unsigned int* row = (const unsigned int*)h2 + (size_t)e * 64;
  #pragma unroll 8
  for (int c = 0; c < 64; ++c) {
    unsigned int p = row[c];
    s += bf2f((unsigned short)(p & 0xFFFFu)) * Wr3[2 * c]
       + bf2f((unsigned short)(p >> 16))     * Wr3[2 * c + 1];
  }
  float sig = 1.f / (1.f + __expf(-s));
  out[e] = 4.f * sig + 1.f;
}

// ---------------- host ----------------
#define NIL8 nullptr, nullptr, nullptr, nullptr, nullptr, nullptr, nullptr, nullptr
extern "C" void kernel_launch(void* const* d_in, const int* in_sizes, int n_in,
                              void* d_out, int out_size, void* d_ws, size_t ws_size,
                              hipStream_t stream) {
  (void)in_sizes; (void)n_in; (void)out_size; (void)ws_size;
  const int*   ei0       = (const int*)d_in[0];
  const int*   ei1       = ((const int*)d_in[0]) + E_N;
  const int*   etype     = (const int*)d_in[1];
  const float* edge_attr = (const float*)d_in[2];
  const float* emb_user  = (const float*)d_in[3];
  const float* emb_item  = (const float*)d_in[4];
  const float* Wj_u = (const float*)d_in[5];  const float* bj_u = (const float*)d_in[6];
  const float* We1_u= (const float*)d_in[7];  const float* be1_u= (const float*)d_in[8];
  const float* We2_u= (const float*)d_in[9];  const float* be2_u= (const float*)d_in[10];
  const float* Wi_u = (const float*)d_in[11]; const float* bi_u = (const float*)d_in[12];
  const float* Wj_i = (const float*)d_in[13]; const float* bj_i = (const float*)d_in[14];
  const float* We1_i= (const float*)d_in[15]; const float* be1_i= (const float*)d_in[16];
  const float* We2_i= (const float*)d_in[17]; const float* be2_i= (const float*)d_in[18];
  const float* Wi_i = (const float*)d_in[19]; const float* bi_i = (const float*)d_in[20];
  const float* Wr1  = (const float*)d_in[21]; const float* br1  = (const float*)d_in[22];
  const float* Wr2  = (const float*)d_in[23]; const float* br2  = (const float*)d_in[24];
  const float* Wr3  = (const float*)d_in[25]; const float* br3  = (const float*)d_in[26];

  // ---- workspace carve ----
  char* p = (char*)d_ws;
  auto carve = [&](size_t bytes) -> void* {
    void* r = (void*)p; p += (bytes + 255) & ~(size_t)255; return r;
  };
  unsigned short* WjT_u  = (unsigned short*)carve((size_t)640 * 128 * 2);
  unsigned short* WjT_i  = (unsigned short*)carve((size_t)640 * 128 * 2);
  unsigned short* We1T_u = (unsigned short*)carve((size_t)640 * 768 * 2);
  unsigned short* We1T_i = (unsigned short*)carve((size_t)640 * 768 * 2);
  unsigned short* We2T_u = (unsigned short*)carve((size_t)128 * 640 * 2);
  unsigned short* We2T_i = (unsigned short*)carve((size_t)128 * 640 * 2);
  unsigned short* WiT_u  = (unsigned short*)carve((size_t)128 * 128 * 2);
  unsigned short* WiT_i  = (unsigned short*)carve((size_t)128 * 128 * 2);
  unsigned short* Wr1T   = (unsigned short*)carve((size_t)128 * 256 * 2);
  unsigned short* Wr2T   = (unsigned short*)carve((size_t)128 * 128 * 2);
  unsigned short* xj_u   = (unsigned short*)carve((size_t)I_NODES * 640 * 2);
  unsigned short* xj_i   = (unsigned short*)carve((size_t)U_N * 640 * 2);
  unsigned short* h1     = (unsigned short*)carve((size_t)E_N * 640 * 2);
  float*          agg_u  = (float*)carve((size_t)U_N * D_DIM * 4);
  float*          agg_i  = (float*)carve((size_t)I_NODES * D_DIM * 4);
  int*            cnt_u  = (int*)carve((size_t)U_N * 4);
  int*            cnt_i  = (int*)carve((size_t)I_NODES * 4);
  unsigned short* xu_c   = (unsigned short*)carve((size_t)U_N * D_DIM * 2);
  unsigned short* xi_c   = (unsigned short*)carve((size_t)I_NODES * D_DIM * 2);
  unsigned short* hh     = h1;                          // reuse h1 region
  unsigned short* h2     = h1 + (size_t)E_N * D_DIM;    // disjoint sub-region

  hipMemsetAsync(agg_u, 0, (size_t)U_N * D_DIM * 4, stream);
  hipMemsetAsync(agg_i, 0, (size_t)I_NODES * D_DIM * 4, stream);
  hipMemsetAsync(cnt_u, 0, (size_t)U_N * 4, stream);
  hipMemsetAsync(cnt_i, 0, (size_t)I_NODES * 4, stream);

  auto tpose = [&](const float* W, unsigned short* WT, int K, int N) {
    transpose_bf16<<<(K * N + 255) / 256, 256, 0, stream>>>(W, WT, K, N);
  };
  tpose(Wj_u, WjT_u, 128, 640);   tpose(Wj_i, WjT_i, 128, 640);
  tpose(We1_u, We1T_u, 768, 640); tpose(We1_i, We1T_i, 768, 640);
  tpose(We2_u, We2T_u, 640, 128); tpose(We2_i, We2T_i, 640, 128);
  tpose(Wi_u, WiT_u, 128, 128);   tpose(Wi_i, WiT_i, 128, 128);
  tpose(Wr1, Wr1T, 256, 128);     tpose(Wr2, Wr2T, 128, 128);

  count_edges<<<(E_N + 255) / 256, 256, 0, stream>>>(ei0, cnt_u, E_N);
  count_edges<<<(E_N + 255) / 256, 256, 0, stream>>>(ei1, cnt_i, E_N);

  const dim3 blk(NTHREADS);
  const dim3 gN5((U_N + BM - 1) / BM, 5);     // node GEMM, N=640
  const dim3 gE5((E_N + BM - 1) / BM, 5);     // edge GEMM1, N=640
  const dim3 gE1((E_N + BM - 1) / BM, 1);     // edge GEMM2 / head, N=128
  const dim3 gN1((U_N + BM - 1) / BM, 1);     // finalize, N=128

  // xj tables (no activation): xj_u = emb_item @ Wj_u ; xj_i = emb_user @ Wj_i
  wmma_gemm<1, 1, false><<<gN5, blk, 0, stream>>>(emb_item, WjT_u, bj_u, xj_u,
      I_NODES, 640, 128, NIL8, nullptr, nullptr, nullptr);
  wmma_gemm<1, 1, false><<<gN5, blk, 0, stream>>>(emb_user, WjT_i, bj_i, xj_i,
      U_N, 640, 128, NIL8, nullptr, nullptr, nullptr);

  // ---- user conv: h1 = relu(edge_attr@We1_u+b), msg->agg_u[ei0], node msgs from xj_u[ei1]
  wmma_gemm<1, 1, true><<<gE5, blk, 0, stream>>>(edge_attr, We1T_u, be1_u, h1,
      E_N, 640, 768, NIL8, nullptr, nullptr, nullptr);
  wmma_gemm<0, 2, true><<<gE1, blk, 0, stream>>>(h1, We2T_u, be2_u, nullptr,
      E_N, 128, 640, nullptr, nullptr, nullptr, nullptr,
      /*s_src*/ ei1, /*s_dst*/ ei0, etype, xj_u, agg_u, nullptr, nullptr);

  // ---- item conv: reuse h1
  wmma_gemm<1, 1, true><<<gE5, blk, 0, stream>>>(edge_attr, We1T_i, be1_i, h1,
      E_N, 640, 768, NIL8, nullptr, nullptr, nullptr);
  wmma_gemm<0, 2, true><<<gE1, blk, 0, stream>>>(h1, We2T_i, be2_i, nullptr,
      E_N, 128, 640, nullptr, nullptr, nullptr, nullptr,
      /*s_src*/ ei0, /*s_dst*/ ei1, etype, xj_i, agg_i, nullptr, nullptr);

  // ---- finalize: x_conved = relu(x_dst@Wi + b + agg/max(cnt,1)) -> bf16
  wmma_gemm<1, 3, false><<<gN1, blk, 0, stream>>>(emb_user, WiT_u, bi_u, xu_c,
      U_N, 128, 128, nullptr, nullptr, nullptr, nullptr,
      nullptr, nullptr, nullptr, nullptr, nullptr, agg_u, cnt_u);
  wmma_gemm<1, 3, false><<<gN1, blk, 0, stream>>>(emb_item, WiT_i, bi_i, xi_c,
      I_NODES, 128, 128, nullptr, nullptr, nullptr, nullptr,
      nullptr, nullptr, nullptr, nullptr, nullptr, agg_i, cnt_i);

  // ---- rating head ----
  wmma_gemm<2, 1, true><<<gE1, blk, 0, stream>>>(nullptr, Wr1T, br1, hh,
      E_N, 128, 256, ei0, ei1, xu_c, xi_c,
      nullptr, nullptr, nullptr, nullptr, nullptr, nullptr, nullptr);
  wmma_gemm<0, 1, true><<<gE1, blk, 0, stream>>>(hh, Wr2T, br2, h2,
      E_N, 128, 128, NIL8, nullptr, nullptr, nullptr);
  head_final<<<(E_N + 255) / 256, 256, 0, stream>>>(h2, Wr3, br3, (float*)d_out, E_N);
}
